// SCAM_80960133529639
// MI455X (gfx1250) — matrix operand order repaired and checked by reading, not compiled, not run on hardware
//
#include <hip/hip_runtime.h>

// SCAM fused kernel for MI455X (gfx1250, wave32, WMMA f16 16x16x32).
// One workgroup (256 threads = 8 waves) per (b,h) scanline. W=320, C=64.

typedef __attribute__((ext_vector_type(16))) _Float16 v16h;
typedef __attribute__((ext_vector_type(8)))  float    v8f;

#define QV_ST 66   // padded LDS row stride (halfs) for 320x64 q/v arrays
#define P_ST  34   // padded LDS row stride (halfs) for 16x32 P staging
#define W_ST  68   // padded LDS row stride (halfs) for 64x64 transposed weights
#define NB    4
#define NH    180
#define NW    320
#define NC    64

__device__ __forceinline__ v8f wmma_f16(v16h a, v16h b, v8f c) {
  // D = A(16x32 f16) * B(32x16 f16) + C(16x16 f32)
  return __builtin_amdgcn_wmma_f32_16x16x32_f16(false, a, false, b, (short)0, c,
                                                false, false);
}

// A-operand (16x32, f16) from row-major LDS [row][col].
// ISA layout: M = lane&15 ; K = (v>=4?16:0) + (lane>=16?8:0) + 2*(v&3) + slot
__device__ __forceinline__ v16h loadA_lds(const _Float16* s, int row0, int k0,
                                          int stride) {
  const int lane = threadIdx.x & 31;
  const _Float16* p = s + (row0 + (lane & 15)) * stride + k0 + ((lane & 16) >> 1);
  v16h a;
#pragma unroll
  for (int v = 0; v < 8; ++v) {
    const int kb = ((v & 4) << 2) | ((v & 3) << 1);
    a[2 * v]     = p[kb];
    a[2 * v + 1] = p[kb + 1];
  }
  return a;
}

// B-operand (32x16, f16) where B = src^T, src row-major [n][k].
// Used both for Q*Q^T (src = other image's q) and for x@W (src = transposed W).
// ISA B layout: N = lane&15 ; K = (i>=8?16:0) + (lane>=16?8:0) + (i&7)
__device__ __forceinline__ v16h loadBT_lds(const _Float16* s, int n0, int k0,
                                           int stride) {
  const int lane = threadIdx.x & 31;
  const _Float16* p = s + (n0 + (lane & 15)) * stride + k0 + ((lane & 16) >> 1);
  v16h b;
#pragma unroll
  for (int i = 0; i < 16; ++i) b[i] = p[(i & 7) + ((i & 8) << 1)];
  return b;
}

// B-operand (32x16, f16) from row-major LDS [k][n] (used for P*V).
__device__ __forceinline__ v16h loadB_lds(const _Float16* s, int k0, int n0,
                                          int stride) {
  const int lane = threadIdx.x & 31;
  const int n = n0 + (lane & 15);
  const int kb = k0 + ((lane & 16) >> 1);
  v16h b;
#pragma unroll
  for (int v = 0; v < 8; ++v) {
    b[2 * v]     = s[(kb + 2 * v + ((v & 4) << 1))     * stride + n];
    b[2 * v + 1] = s[(kb + 2 * v + ((v & 4) << 1) + 1) * stride + n];
  }
  return b;
}

// One-time per block: convert 64x64 f32 weight to f16 LDS, transposed [n][k].
// Global reads fully coalesced; LDS b16 writes land in distinct banks (W_ST=68).
__device__ __forceinline__ void stage_weight(const float* __restrict__ w,
                                             _Float16* dst) {
  for (int e = threadIdx.x; e < NC * NC; e += 256) {
    const int k = e >> 6, n = e & 63;
    dst[n * W_ST + k] = (_Float16)w[e];
  }
}

// Projection of one 16-row tile: dst = [LN](x) @ W + bias, stored f16 in LDS.
// wT = transposed f16 weight in LDS ([c_out][c_in], stride W_ST).
__device__ __forceinline__ void proj_tile(const float* __restrict__ x,
                                          const _Float16* wT,
                                          const float* __restrict__ bias,
                                          const float* __restrict__ lns,
                                          const float* __restrict__ lnb,
                                          _Float16* dst, int t, bool doLN) {
  const int lane = threadIdx.x & 31;
  const int hi8 = (lane & 16) >> 1;          // +8 rows / +8 K for upper half-wave
  const float* xp = x + (t * 16 + (lane & 15)) * NC;

  // Load this lane's 32 A-fragment values (2 K-fragments) as f32.
  float xv[32];
#pragma unroll
  for (int f = 0; f < 2; ++f)
#pragma unroll
    for (int v = 0; v < 8; ++v) {
      const int k = f * 32 + hi8 + (((v & 4) << 2) | ((v & 3) << 1));
      xv[f * 16 + 2 * v]     = xp[k];
      xv[f * 16 + 2 * v + 1] = xp[k + 1];
    }

  v16h a0, a1;
  if (doLN) {
    float s = 0.f, s2 = 0.f;
#pragma unroll
    for (int i = 0; i < 32; ++i) { s += xv[i]; s2 += xv[i] * xv[i]; }
    s  += __shfl_xor(s, 16);                 // lane^16 holds the row's other K half
    s2 += __shfl_xor(s2, 16);
    const float mu  = s * (1.0f / 64.0f);
    const float var = s2 * (1.0f / 64.0f) - mu * mu;
    const float rs  = rsqrtf(var + 1e-6f);
#pragma unroll
    for (int f = 0; f < 2; ++f)
#pragma unroll
      for (int v = 0; v < 8; ++v)
#pragma unroll
        for (int sl = 0; sl < 2; ++sl) {
          const int k = f * 32 + hi8 + (((v & 4) << 2) | ((v & 3) << 1)) + sl;
          const float nv = (xv[f * 16 + 2 * v + sl] - mu) * rs * lns[k] + lnb[k];
          if (f == 0) a0[2 * v + sl] = (_Float16)nv;
          else        a1[2 * v + sl] = (_Float16)nv;
        }
  } else {
#pragma unroll
    for (int i = 0; i < 16; ++i) {
      a0[i] = (_Float16)xv[i];
      a1[i] = (_Float16)xv[16 + i];
    }
  }

#pragma unroll
  for (int ct = 0; ct < 4; ++ct) {
    v8f d = {0.f, 0.f, 0.f, 0.f, 0.f, 0.f, 0.f, 0.f};
    d = wmma_f16(a0, loadBT_lds(wT, ct * 16, 0,  W_ST), d);
    d = wmma_f16(a1, loadBT_lds(wT, ct * 16, 32, W_ST), d);
    const float bb = bias[ct * 16 + (lane & 15)];
#pragma unroll
    for (int v = 0; v < 8; ++v)
      dst[(t * 16 + v + hi8) * QV_ST + ct * 16 + (lane & 15)] =
          (_Float16)(d[v] + bb);
  }
}

// One 16-row output strip of flash-attention: rows of sQa attend over all 320
// rows of sQb, weighted sum of sV; out = xres + (softmax(S) @ V) * mult.
__device__ __forceinline__ void attend_tile(const _Float16* sQa,
                                            const _Float16* sQb,
                                            const _Float16* sV, _Float16* sPw,
                                            const float* __restrict__ mult,
                                            const float* __restrict__ xres,
                                            float* __restrict__ outp, int t) {
  const int lane = threadIdx.x & 31;
  const int hi8 = (lane & 16) >> 1;
  const v16h aq0 = loadA_lds(sQa, t * 16, 0,  QV_ST);
  const v16h aq1 = loadA_lds(sQa, t * 16, 32, QV_ST);

  float m[8], ssum[8];
  v8f acc[4];
#pragma unroll
  for (int v = 0; v < 8; ++v) { m[v] = -3.0e38f; ssum[v] = 0.0f; }
#pragma unroll
  for (int ct = 0; ct < 4; ++ct) acc[ct] = {0.f, 0.f, 0.f, 0.f, 0.f, 0.f, 0.f, 0.f};

  for (int r0 = 0; r0 < NW; r0 += 32) {
    const v8f z = {0.f, 0.f, 0.f, 0.f, 0.f, 0.f, 0.f, 0.f};
    // Two 16x16 score tiles: S = qA @ qB^T (K = 64 -> two k-steps)
    v8f s0 = wmma_f16(aq0, loadBT_lds(sQb, r0,      0,  QV_ST), z);
    s0     = wmma_f16(aq1, loadBT_lds(sQb, r0,      32, QV_ST), s0);
    v8f s1 = wmma_f16(aq0, loadBT_lds(sQb, r0 + 16, 0,  QV_ST), z);
    s1     = wmma_f16(aq1, loadBT_lds(sQb, r0 + 16, 32, QV_ST), s1);

    float nm[8];
#pragma unroll
    for (int v = 0; v < 8; ++v) {
      s0[v] *= 0.125f;
      s1[v] *= 0.125f;
      nm[v] = fmaxf(s0[v], s1[v]);
    }
    // Row max across the 16 columns (lanes 0-15 / 16-31 are separate halves).
#pragma unroll
    for (int off = 1; off <= 8; off <<= 1)
#pragma unroll
      for (int v = 0; v < 8; ++v)
        nm[v] = fmaxf(nm[v], __shfl_xor(nm[v], off));

#pragma unroll
    for (int v = 0; v < 8; ++v) {
      nm[v] = fmaxf(m[v], nm[v]);
      const float rf = __expf(m[v] - nm[v]);
      m[v] = nm[v];
      ssum[v] *= rf;
      acc[0][v] *= rf; acc[1][v] *= rf; acc[2][v] *= rf; acc[3][v] *= rf;
      const float p0 = __expf(s0[v] - nm[v]);
      const float p1 = __expf(s1[v] - nm[v]);
      ssum[v] += p0 + p1;                       // per-lane partial row sum
      const int Mr = v + hi8;                   // stage P (D-layout -> LDS)
      sPw[Mr * P_ST + (lane & 15)]      = (_Float16)p0;
      sPw[Mr * P_ST + 16 + (lane & 15)] = (_Float16)p1;
    }
    // Same-wave LDS ops complete in order on CDNA5; block compiler reordering.
    asm volatile("" ::: "memory");
    const v16h pa = loadA_lds(sPw, 0, 0, P_ST); // P as A-operand (16x32)
    acc[0] = wmma_f16(pa, loadB_lds(sV, r0, 0,  QV_ST), acc[0]);
    acc[1] = wmma_f16(pa, loadB_lds(sV, r0, 16, QV_ST), acc[1]);
    acc[2] = wmma_f16(pa, loadB_lds(sV, r0, 32, QV_ST), acc[2]);
    acc[3] = wmma_f16(pa, loadB_lds(sV, r0, 48, QV_ST), acc[3]);
    asm volatile("" ::: "memory");
  }

  // Finish the row sums across the 16 lanes of each half.
#pragma unroll
  for (int off = 1; off <= 8; off <<= 1)
#pragma unroll
    for (int v = 0; v < 8; ++v)
      ssum[v] += __shfl_xor(ssum[v], off);

  const int col = lane & 15;
#pragma unroll
  for (int ct = 0; ct < 4; ++ct) {
    const float mu = mult[ct * 16 + col];
#pragma unroll
    for (int v = 0; v < 8; ++v) {
      const int row = t * 16 + v + hi8;
      const int idx = row * NC + ct * 16 + col;
      outp[idx] = xres[idx] + (acc[ct][v] / ssum[v]) * mu;
    }
  }
}

__global__ __launch_bounds__(256, 1) void scam_kernel(
    const float* __restrict__ x_l, const float* __restrict__ x_r,
    const float* __restrict__ ln_l_s, const float* __restrict__ ln_l_b,
    const float* __restrict__ ln_r_s, const float* __restrict__ ln_r_b,
    const float* __restrict__ w_ql, const float* __restrict__ b_ql,
    const float* __restrict__ w_qr, const float* __restrict__ b_qr,
    const float* __restrict__ w_vl, const float* __restrict__ b_vl,
    const float* __restrict__ w_vr, const float* __restrict__ b_vr,
    const float* __restrict__ beta, const float* __restrict__ gamma,
    float* __restrict__ out) {
  __shared__ _Float16 sQa[NW * QV_ST];      // q_l
  __shared__ _Float16 sQb[NW * QV_ST];      // q_r
  __shared__ _Float16 sV [NW * QV_ST];      // v_r, then reused for v_l
  __shared__ _Float16 sW [3 * NC * W_ST];   // w_ql^T, w_qr^T, w_vr^T (-> w_vl^T)
  __shared__ _Float16 sP [8 * 16 * P_ST];   // per-wave P staging

  const int bh = blockIdx.x;
  const float* xl = x_l + (size_t)bh * NW * NC;
  const float* xr = x_r + (size_t)bh * NW * NC;
  float* outl = out + (size_t)bh * NW * NC;
  float* outr = out + (size_t)NB * NH * NW * NC + (size_t)bh * NW * NC;
  const int wave = threadIdx.x >> 5;
  _Float16* sPw = sP + wave * 16 * P_ST;
  _Float16* sWql = sW;
  _Float16* sWqr = sW + NC * W_ST;
  _Float16* sWv  = sW + 2 * NC * W_ST;

  // Phase 0: stage weights (f32 -> f16, transposed) into LDS once per block.
  stage_weight(w_ql, sWql);
  stage_weight(w_qr, sWqr);
  stage_weight(w_vr, sWv);
  __syncthreads();

  // Phase A: q_l, q_r, v_r projections (60 tile-jobs across 8 waves).
  for (int job = wave; job < 60; job += 8) {
    const int kind = job / 20, t = job % 20;
    if (kind == 0)      proj_tile(xl, sWql, b_ql, ln_l_s, ln_l_b, sQa, t, true);
    else if (kind == 1) proj_tile(xr, sWqr, b_qr, ln_r_s, ln_r_b, sQb, t, true);
    else                proj_tile(xr, sWv,  b_vr, nullptr, nullptr, sV, t, false);
  }
  __syncthreads();

  // w_vr^T is dead now (v_r values live in sV); refill the slot with w_vl^T.
  // A __syncthreads separates this from phase A2's reads.
  stage_weight(w_vl, sWv);

  // Direction 1 (r -> l): row softmax of S, weighted v_r, scaled by beta.
  for (int t = wave; t < 20; t += 8)
    attend_tile(sQa, sQb, sV, sPw, beta, xl, outl, t);
  __syncthreads();

  // Phase A2: recompute v_l into the (now free) sV buffer.
  for (int t = wave; t < 20; t += 8)
    proj_tile(xl, sWv, b_vl, nullptr, nullptr, sV, t, false);
  __syncthreads();

  // Direction 2 (l -> r): column softmax == row softmax of S^T; v_l, gamma.
  for (int t = wave; t < 20; t += 8)
    attend_tile(sQb, sQa, sV, sPw, gamma, xr, outr, t);
}

extern "C" void kernel_launch(void* const* d_in, const int* in_sizes, int n_in,
                              void* d_out, int out_size, void* d_ws,
                              size_t ws_size, hipStream_t stream) {
  (void)in_sizes; (void)n_in; (void)out_size; (void)d_ws; (void)ws_size;
  const float* x_l    = (const float*)d_in[0];
  const float* x_r    = (const float*)d_in[1];
  const float* ln_l_s = (const float*)d_in[2];
  const float* ln_l_b = (const float*)d_in[3];
  const float* ln_r_s = (const float*)d_in[4];
  const float* ln_r_b = (const float*)d_in[5];
  const float* w_ql   = (const float*)d_in[6];
  const float* b_ql   = (const float*)d_in[7];
  const float* w_qr   = (const float*)d_in[8];
  const float* b_qr   = (const float*)d_in[9];
  const float* w_vl   = (const float*)d_in[10];
  const float* b_vl   = (const float*)d_in[11];
  const float* w_vr   = (const float*)d_in[12];
  const float* b_vr   = (const float*)d_in[13];
  const float* beta   = (const float*)d_in[14];
  const float* gamma  = (const float*)d_in[15];
  float* out = (float*)d_out;

  scam_kernel<<<dim3(NB * NH), dim3(256), 0, stream>>>(
      x_l, x_r, ln_l_s, ln_l_b, ln_r_s, ln_r_b, w_ql, b_ql, w_qr, b_qr, w_vl,
      b_vl, w_vr, b_vr, beta, gamma, out);
}
